// GATLayer_15582141350268
// MI455X (gfx1250) — compile-verified, matrix operand order checked by reading, not run on hardware
//
#include <hip/hip_runtime.h>
#include <hip/hip_bf16.h>

// ---- CDNA5 WMMA types -------------------------------------------------------
typedef __attribute__((ext_vector_type(16))) __bf16 v16bf;
typedef __attribute__((ext_vector_type(8)))  float  v8f;

union BF16x16 { uint4 u[2]; v16bf v; unsigned short s[16]; };

__device__ __forceinline__ unsigned short bf16_bits(float f) {
  __bf16 h = (__bf16)f;                       // RNE convert
  return __builtin_bit_cast(unsigned short, h);
}
__device__ __forceinline__ unsigned pack_bf16(float lo, float hi) {
  return (unsigned)bf16_bits(lo) | ((unsigned)bf16_bits(hi) << 16);
}

// Geometry
#define GAT_N     12
#define GAT_F     256
#define GAT_D     256
#define BPB       4          // batch items per block
#define ROWS      48         // BPB * GAT_N  (== 3 WMMA m-tiles, no padding)
#define SH        260        // Hs f32 row stride (bank-stagger)
#define SX        264        // xb bf16 row stride (528B rows -> optimal b128 LDS loads)
#define NEG_SLOPE 0.2f

// Shared memory carve-up (xb aliases Hs; xb is dead before Hs is written)
#define HS_BYTES    (ROWS * SH * 4)            // 49920
#define AS_OFF      HS_BYTES                   // a vector, 512 f32
#define SDST_OFF    (AS_OFF + 2048)            // 48 f32
#define SSRC_OFF    (SDST_OFF + 192)           // 48 f32
#define ALPHA_OFF   (SSRC_OFF + 192)           // 48*12 f32
#define SMEM_BYTES  (ALPHA_OFF + ROWS * GAT_N * 4)   // 54656

// ---- Kernel 0: pack W (f32, row-major KxN) into bf16 B-fragments ------------
// Fragment (nt,k) covers cols [nt*16,nt*16+16), K rows [k*32,k*32+32).
// Lane l<16  -> col nt*16+l,    K = k*32 + {0..7, 16..23}
// Lane l>=16 -> col nt*16+l-16, K = k*32 + {8..15, 24..31}
// Stored 32B contiguous per lane so the GEMM loads it with 2x b128.
__global__ void gat_wfrag_build(const float* __restrict__ W,
                                unsigned short* __restrict__ wfrag) {
  const int frag = blockIdx.x;           // 0..127  (nt*8 + k)
  const int lane = threadIdx.x;          // 0..31
  const int nt = frag >> 3, k = frag & 7;
  const int col = nt * 16 + (lane & 15);
  const int kb  = k * 32 + (lane >> 4) * 8;
  BF16x16 ub;
#pragma unroll
  for (int i = 0; i < 8; ++i) {
    ub.s[i]     = bf16_bits(W[(kb + i) * GAT_D + col]);
    ub.s[8 + i] = bf16_bits(W[(kb + 16 + i) * GAT_D + col]);
  }
  uint4* dst = (uint4*)(wfrag + (size_t)(frag * 32 + lane) * 16);
  dst[0] = ub.u[0];
  dst[1] = ub.u[1];
}

// ---- Fused GAT layer --------------------------------------------------------
template <bool FRAG>
__global__ __launch_bounds__(256) void gat_fused(
    const float* __restrict__ x, const int* __restrict__ adj,
    const float* __restrict__ W, const float* __restrict__ a,
    const unsigned short* __restrict__ wfrag, float* __restrict__ out) {
  __shared__ __align__(16) char smem[SMEM_BYTES];
  float*  Hs    = (float*)smem;                  // ROWS x SH f32 (after GEMM)
  __bf16* xb    = (__bf16*)smem;                 // ROWS x SX bf16 (before GEMM)
  float*  a_s   = (float*)(smem + AS_OFF);
  float*  sdst  = (float*)(smem + SDST_OFF);
  float*  ssrc  = (float*)(smem + SSRC_OFF);
  float*  alpha = (float*)(smem + ALPHA_OFF);

  const int tid  = threadIdx.x;
  const int wv   = tid >> 5;        // wave 0..7, owns cols [wv*32, wv*32+32)
  const int lane = tid & 31;
  const int lhi  = lane >> 4;
  const int llo  = lane & 15;

  // ---- Phase 1: stage x tile (48x256 f32 -> bf16 LDS) + a vector ----
  {
    const float4* xg = (const float4*)(x + (size_t)blockIdx.x * ROWS * GAT_F);
#pragma unroll
    for (int it = 0; it < 12; ++it) {
      const int fi  = tid + 256 * it;          // 0..3071 float4s
      const int row = fi >> 6;                 // 64 float4 per row
      const int c4  = (fi & 63) << 2;
      const float4 v = xg[fi];
      *(unsigned*)&xb[row * SX + c4]     = pack_bf16(v.x, v.y);
      *(unsigned*)&xb[row * SX + c4 + 2] = pack_bf16(v.z, v.w);
    }
    a_s[tid]       = a[tid];
    a_s[256 + tid] = a[256 + tid];
  }
  __syncthreads();

  // ---- Phase 2: H = x @ W via v_wmma_f32_16x16x32_bf16 ----
  v8f acc[3][2];
#pragma unroll
  for (int mt = 0; mt < 3; ++mt)
#pragma unroll
    for (int nt2 = 0; nt2 < 2; ++nt2)
      acc[mt][nt2] = (v8f){0.f, 0.f, 0.f, 0.f, 0.f, 0.f, 0.f, 0.f};

#pragma unroll
  for (int k = 0; k < 8; ++k) {
    v16bf Bf[2];
#pragma unroll
    for (int nt2 = 0; nt2 < 2; ++nt2) {
      BF16x16 ub;
      if (FRAG) {
        const uint4* p = (const uint4*)(wfrag +
            (size_t)((((wv * 2 + nt2) * 8 + k) * 32 + lane)) * 16);
        ub.u[0] = p[0];
        ub.u[1] = p[1];
      } else {
        const int col = (wv * 2 + nt2) * 16 + llo;
        const int kb  = k * 32 + lhi * 8;
#pragma unroll
        for (int i = 0; i < 8; ++i) {
          ub.s[i]     = bf16_bits(W[(kb + i) * GAT_D + col]);
          ub.s[8 + i] = bf16_bits(W[(kb + 16 + i) * GAT_D + col]);
        }
      }
      Bf[nt2] = ub.v;
    }
#pragma unroll
    for (int mt = 0; mt < 3; ++mt) {
      const int row = mt * 16 + llo;
      const int kb  = k * 32 + lhi * 8;
      BF16x16 ua;
      ua.u[0] = *(const uint4*)&xb[row * SX + kb];
      ua.u[1] = *(const uint4*)&xb[row * SX + kb + 16];
      const v16bf Af = ua.v;
      acc[mt][0] = __builtin_amdgcn_wmma_f32_16x16x32_bf16(
          false, Af, false, Bf[0], (short)0, acc[mt][0], false, false);
      acc[mt][1] = __builtin_amdgcn_wmma_f32_16x16x32_bf16(
          false, Af, false, Bf[1], (short)0, acc[mt][1], false, false);
    }
  }
  __syncthreads();   // xb dead from here; Hs may overwrite it

  // C/D layout: vgpr r, lanes 0-15 -> (M=r, N=lane); lanes 16-31 -> (M=8+r)
#pragma unroll
  for (int mt = 0; mt < 3; ++mt)
#pragma unroll
    for (int nt2 = 0; nt2 < 2; ++nt2) {
      const int col = wv * 32 + nt2 * 16 + llo;
#pragma unroll
      for (int r = 0; r < 8; ++r)
        Hs[(mt * 16 + lhi * 8 + r) * SH + col] = acc[mt][nt2][r];
    }
  __syncthreads();

  // ---- Phase 3: s_dst = H.a[:D], s_src = H.a[D:]  (8 lanes per row) ----
  {
    const int g  = tid >> 3;    // group 0..31
    const int l8 = tid & 7;
#pragma unroll
    for (int p = 0; p < 2; ++p) {
      const int row = p * 32 + g;
      if (row < ROWS) {
        const float* hrow = &Hs[row * SH];
        const int e0 = l8 * 32;
        float s0 = 0.f, s1 = 0.f;
#pragma unroll
        for (int i = 0; i < 32; ++i) {
          const float h = hrow[e0 + i];
          s0 = fmaf(h, a_s[e0 + i], s0);
          s1 = fmaf(h, a_s[256 + e0 + i], s1);
        }
#pragma unroll
        for (int m = 1; m < 8; m <<= 1) {
          s0 += __shfl_xor(s0, m, 32);
          s1 += __shfl_xor(s1, m, 32);
        }
        if (l8 == 0) { sdst[row] = s0; ssrc[row] = s1; }
      }
    }
  }
  __syncthreads();

  // ---- Phase 4: masked leaky-relu logits + softmax over j (12) ----
  if (tid < ROWS) {
    const int bl = tid / GAT_N;
    const int i  = tid % GAT_N;
    const float si = ssrc[tid];
    float lg[GAT_N];
    float mx = -3.0e38f;
#pragma unroll
    for (int j = 0; j < GAT_N; ++j) {
      float e = si + sdst[bl * GAT_N + j];
      e = e > 0.f ? e : NEG_SLOPE * e;
      e = (adj[i * GAT_N + j] == 1) ? e : 0.f;   // masked logits are 0, not -inf
      lg[j] = e;
      mx = fmaxf(mx, e);
    }
    float sum = 0.f;
#pragma unroll
    for (int j = 0; j < GAT_N; ++j) {
      const float ex = expf(lg[j] - mx);
      lg[j] = ex;
      sum += ex;
    }
    const float inv = 1.f / sum;
#pragma unroll
    for (int j = 0; j < GAT_N; ++j) alpha[tid * GAT_N + j] = lg[j] * inv;
  }
  __syncthreads();

  // ---- Phase 5: out = elu(alpha @ H), float2 per thread ----
  {
    const int c2   = (tid & 127) * 2;
    const int half = tid >> 7;
    for (int r = half * 24; r < half * 24 + 24; ++r) {
      const int bl = r / GAT_N;
      const float* ar = &alpha[r * GAT_N];
      float ax = 0.f, ay = 0.f;
#pragma unroll
      for (int j = 0; j < GAT_N; ++j) {
        const float aw = ar[j];
        const float2 h = *(const float2*)&Hs[(bl * GAT_N + j) * SH + c2];
        ax = fmaf(aw, h.x, ax);
        ay = fmaf(aw, h.y, ay);
      }
      ax = ax > 0.f ? ax : (expf(ax) - 1.f);
      ay = ay > 0.f ? ay : (expf(ay) - 1.f);
      const size_t grow = (size_t)blockIdx.x * ROWS + r;
      *(float2*)&out[grow * GAT_D + c2] = make_float2(ax, ay);
    }
  }
}

// ---- Launcher ---------------------------------------------------------------
extern "C" void kernel_launch(void* const* d_in, const int* in_sizes, int n_in,
                              void* d_out, int out_size, void* d_ws, size_t ws_size,
                              hipStream_t stream) {
  const float* x   = (const float*)d_in[0];
  const int*   adj = (const int*)d_in[1];
  const float* W   = (const float*)d_in[2];
  const float* a   = (const float*)d_in[3];
  float* out = (float*)d_out;

  const int B      = in_sizes[0] / (GAT_N * GAT_F);   // 16384
  const int blocks = B / BPB;                          // 4096

  const size_t wfrag_bytes = (size_t)GAT_F * GAT_D * 2;  // 128 KB
  if (d_ws && ws_size >= wfrag_bytes) {
    unsigned short* wfrag = (unsigned short*)d_ws;
    gat_wfrag_build<<<128, 32, 0, stream>>>(W, wfrag);
    gat_fused<true><<<blocks, 256, 0, stream>>>(x, adj, W, a, wfrag, out);
  } else {
    gat_fused<false><<<blocks, 256, 0, stream>>>(x, adj, W, a, nullptr, out);
  }
}
